// RNNDecoder_36335423324772
// MI455X (gfx1250) — matrix operand order repaired and checked by reading, not compiled, not run on hardware
//
#include <hip/hip_runtime.h>
#include <math.h>

typedef __attribute__((ext_vector_type(16))) _Float16 v16h;
typedef __attribute__((ext_vector_type(8)))  _Float16 v8h;
typedef __attribute__((ext_vector_type(8)))  float    v8f;

#define LATENT 256
#define HIDDEN 512
#define MAXN   64
#define BATCH  128
#define NPAIR  2016   // 64 choose 2
#define SEQ    (BATCH * MAXN)   // 8192 rows

// ---------------------------------------------------------------------------
// Elementwise helpers
// ---------------------------------------------------------------------------
__global__ void k_convert(const float* __restrict__ s, _Float16* __restrict__ d, int n) {
    int i = blockIdx.x * blockDim.x + threadIdx.x;
    if (i < n) d[i] = (_Float16)s[i];
}

// split adj_W1 (512 x 1024, row-major) into W1a = [:, :512], W1b = [:, 512:]
__global__ void k_split_w1(const float* __restrict__ s, _Float16* __restrict__ a,
                           _Float16* __restrict__ b) {
    int i = blockIdx.x * blockDim.x + threadIdx.x;
    if (i >= 512 * 512) return;
    int r = i >> 9, c = i & 511;
    a[i] = (_Float16)s[r * 1024 + c];
    b[i] = (_Float16)s[r * 1024 + 512 + c];
}

__global__ void k_zero_u32(unsigned int* __restrict__ p, int n) {
    int i = blockIdx.x * blockDim.x + threadIdx.x;
    if (i < n) p[i] = 0u;
}

// A0[b*64+t][k] = k<256 ? z[b][k]+pe[t][k] : emb[t][k-256]   (f16, 8192 x 512)
__global__ void k_build_a0(const float* __restrict__ z, const float* __restrict__ pe,
                           const float* __restrict__ emb, _Float16* __restrict__ a0) {
    int i = blockIdx.x * blockDim.x + threadIdx.x;
    if (i >= SEQ * 512) return;
    int k = i & 511;
    int t = (i >> 9) & 63;
    int b = i >> 15;
    float v = (k < 256) ? (z[b * 256 + k] + pe[t * 256 + k]) : emb[t * 256 + (k - 256)];
    a0[i] = (_Float16)v;
}

// seq_h = mask(t < n_nodes[b]) ? f16(seq_f32) : 0     (8192 x 256)
__global__ void k_mask_seq(const float* __restrict__ s, _Float16* __restrict__ d,
                           const int* __restrict__ n_nodes) {
    int i = blockIdx.x * blockDim.x + threadIdx.x;
    if (i >= SEQ * 256) return;
    int t = (i >> 8) & 63;
    int b = i >> 14;
    d[i] = (t < n_nodes[b]) ? (_Float16)s[i] : (_Float16)0.0f;
}

// ---------------------------------------------------------------------------
// Register-tiled WMMA GEMM:
//   C(MxN) = A(MxK, f16 row-major) * B(NxK, f16 row-major)^T + bias
// One wave = one (16*MT) x (16*NT) output strip: MT A-frags x NT B-frags per
// K-step of 32, MT*NT v_wmma_f32_16x16x32_f16 per step. Frag loads follow the
// ISA 7.12.2 16-bit layouts (two 16B loads per fragment).
//   <2,4>: 12 loads -> 8 WMMAs per K-step (big batched GEMMs)
//   <1,1>:  4 loads -> 1 WMMA  (latency-critical per-timestep GRU GEMM)
// ---------------------------------------------------------------------------
template <int MT, int NT>
__global__ __launch_bounds__(128) void k_wmma_gemm(
    const _Float16* __restrict__ A, const _Float16* __restrict__ B,
    const float* __restrict__ bias, void* __restrict__ C,
    int M, int N, int K, int store_half)
{
    int wave = threadIdx.x >> 5;
    int lane = threadIdx.x & 31;
    int tn_cnt = N / (16 * NT);
    int total  = (M / (16 * MT)) * tn_cnt;
    int tile = blockIdx.x * 4 + wave;
    if (tile >= total) return;               // wave-uniform: EXEC stays all-ones
    int tm = tile / tn_cnt;
    int tn = tile - tm * tn_cnt;

    int half = lane >> 4;                    // lane group (0: lanes 0-15, 1: 16-31)
    int l    = lane & 15;

    const _Float16* Arow[MT];
    const _Float16* Brow[NT];
    float bv[NT];
    #pragma unroll
    for (int m = 0; m < MT; ++m)
        Arow[m] = A + (size_t)(tm * 16 * MT + m * 16 + l) * K;
    #pragma unroll
    for (int n = 0; n < NT; ++n) {
        Brow[n] = B + (size_t)(tn * 16 * NT + n * 16 + l) * K;
        bv[n] = bias ? bias[tn * 16 * NT + n * 16 + l] : 0.0f;
    }

    v8f acc[MT][NT];
    #pragma unroll
    for (int m = 0; m < MT; ++m)
        #pragma unroll
        for (int n = 0; n < NT; ++n)
            acc[m][n] = (v8f){0.f, 0.f, 0.f, 0.f, 0.f, 0.f, 0.f, 0.f};

    for (int k0 = 0; k0 < K; k0 += 32) {
        v16h av[MT], bw[NT];
        // A 16-bit 16x32 layout: lanes<16 hold K {0..7,16..23}, lanes>=16 {8..15,24..31}
        #pragma unroll
        for (int m = 0; m < MT; ++m) {
            v8h a0 = *(const v8h*)(Arow[m] + k0 + half * 8);
            v8h a1 = *(const v8h*)(Arow[m] + k0 + 16 + half * 8);
            #pragma unroll
            for (int e = 0; e < 8; ++e) { av[m][e] = a0[e]; av[m][e + 8] = a1[e]; }
        }
        // B 16-bit 32x16 layout: lanes<16 hold K {0..7,16..23} of their column, lanes>=16 {8..15,24..31}
        #pragma unroll
        for (int n = 0; n < NT; ++n) {
            v8h b0 = *(const v8h*)(Brow[n] + k0 + half * 8);
            v8h b1 = *(const v8h*)(Brow[n] + k0 + 16 + half * 8);
            #pragma unroll
            for (int e = 0; e < 8; ++e) { bw[n][e] = b0[e]; bw[n][e + 8] = b1[e]; }
        }
        __builtin_prefetch(Arow[0] + k0 + 128, 0, 3);    // next K slab, near-cache
        __builtin_prefetch(Brow[0] + k0 + 128, 0, 3);

        #pragma unroll
        for (int m = 0; m < MT; ++m)
            #pragma unroll
            for (int n = 0; n < NT; ++n)
                acc[m][n] = __builtin_amdgcn_wmma_f32_16x16x32_f16(
                    false, av[m], false, bw[n], (short)0, acc[m][n], false, false);
    }

    // C 32-bit 16x16 layout: col = base + (lane&15); VGPR v -> row base + half*8 + v
    #pragma unroll
    for (int n = 0; n < NT; ++n) {
        int col = tn * 16 * NT + n * 16 + l;
        #pragma unroll
        for (int m = 0; m < MT; ++m) {
            int rbase = tm * 16 * MT + m * 16 + half * 8;
            if (store_half) {
                _Float16* Ch = (_Float16*)C;
                #pragma unroll
                for (int v = 0; v < 8; ++v)
                    Ch[(size_t)(rbase + v) * N + col] = (_Float16)(acc[m][n][v] + bv[n]);
            } else {
                float* Cf = (float*)C;
                #pragma unroll
                for (int v = 0; v < 8; ++v)
                    Cf[(size_t)(rbase + v) * N + col] = acc[m][n][v] + bv[n];
            }
        }
    }
}

// ---------------------------------------------------------------------------
// GRU gate fusion for one timestep.
// gi: (8192 x 1536) with bih already added (GEMM bias); gh: (128 x 1536) with bhh.
// Updates h (f32 + f16 copies) and writes layer output row t (optionally masked).
// ---------------------------------------------------------------------------
__device__ __forceinline__ float sigf(float x) { return 1.0f / (1.0f + __expf(-x)); }

__global__ void k_gru_gate(const float* __restrict__ gi, const float* __restrict__ gh,
                           float* __restrict__ h, _Float16* __restrict__ h_h,
                           _Float16* __restrict__ out_h, const int* __restrict__ n_nodes,
                           int t, int apply_mask)
{
    int i = blockIdx.x * blockDim.x + threadIdx.x;      // 128*512
    if (i >= BATCH * HIDDEN) return;
    int b = i >> 9, j = i & 511;
    size_t grow = (size_t)(b * 64 + t) * 1536;
    size_t hrow = (size_t)b * 1536;
    float ir = gi[grow + j], iz = gi[grow + 512 + j], in = gi[grow + 1024 + j];
    float hr = gh[hrow + j], hz = gh[hrow + 512 + j], hn = gh[hrow + 1024 + j];
    float r  = sigf(ir + hr);
    float zg = sigf(iz + hz);
    float nn = tanhf(in + r * hn);
    float hnew = (1.0f - zg) * nn + zg * h[i];
    h[i]   = hnew;
    h_h[i] = (_Float16)hnew;
    float o = hnew;
    if (apply_mask && t >= n_nodes[b]) o = 0.0f;
    out_h[(size_t)(b * 64 + t) * 512 + j] = (_Float16)o;
}

// ---------------------------------------------------------------------------
// Fused pair head: one wave per (batch, pair).
// hmid = relu(U[b,i] + V[b,j])   (b1 folded into V), logits = hmid @ W2^T + b2,
// Gumbel argmax over 2 classes -> binary adj entry, symmetric scatter + node mask.
// ---------------------------------------------------------------------------
__global__ __launch_bounds__(128) void k_pair_adj(
    const float* __restrict__ U, const float* __restrict__ V,
    const float* __restrict__ W2, const float* __restrict__ b2,
    const float* __restrict__ gumbel, const int* __restrict__ n_nodes,
    float* __restrict__ out)
{
    int wg = blockIdx.x * 4 + (threadIdx.x >> 5);
    if (wg >= BATCH * NPAIR) return;
    int lane = threadIdx.x & 31;
    int b = wg / NPAIR;
    int p = wg - b * NPAIR;

    // invert triu(k=1) flat index: S(i) = (127i - i^2)/2, S(i) <= p < S(i+1)
    int i = (int)((127.0f - sqrtf(16129.0f - 8.0f * (float)p)) * 0.5f);
    if (i < 0) i = 0;
    if (i > 62) i = 62;
    while (i > 0 && ((127 * i - i * i) >> 1) > p) --i;
    while (i < 62 && ((127 * (i + 1) - (i + 1) * (i + 1)) >> 1) <= p) ++i;
    int j = i + 1 + (p - ((127 * i - i * i) >> 1));

    const float* u = U + (size_t)(b * 64 + i) * 512;
    const float* v = V + (size_t)(b * 64 + j) * 512;
    float a0 = 0.f, a1 = 0.f;
    for (int k = lane; k < 512; k += 32) {
        float hm = u[k] + v[k];                  // b1 already inside V
        hm = hm > 0.f ? hm : 0.f;                // relu
        a0 += hm * W2[k];
        a1 += hm * W2[512 + k];
    }
    #pragma unroll
    for (int off = 16; off > 0; off >>= 1) {     // wave32 reduction
        a0 += __shfl_down(a0, off, 32);
        a1 += __shfl_down(a1, off, 32);
    }
    if (lane == 0) {
        float l0 = a0 + b2[0], l1 = a1 + b2[1];
        size_t g = (size_t)(b * NPAIR + p) * 2;
        float g0 = -logf(-logf(gumbel[g]     + 1e-10f) + 1e-10f);
        float g1 = -logf(-logf(gumbel[g + 1] + 1e-10f) + 1e-10f);
        float val = (l0 + g0) >= (l1 + g1) ? 1.0f : 0.0f;   // argmax -> one-hot[0]
        int n = n_nodes[b];
        if (i >= n || j >= n) val = 0.0f;
        out[(size_t)b * 4096 + i * 64 + j] = val;
        out[(size_t)b * 4096 + j * 64 + i] = val;
    }
}

// ---------------------------------------------------------------------------
// Host-side orchestration
// ---------------------------------------------------------------------------
static inline int cdiv(int a, int b) { return (a + b - 1) / b; }

extern "C" void kernel_launch(void* const* d_in, const int* in_sizes, int n_in,
                              void* d_out, int out_size, void* d_ws, size_t ws_size,
                              hipStream_t stream)
{
    (void)in_sizes; (void)n_in; (void)out_size; (void)ws_size;

    const float* z       = (const float*)d_in[0];
    const int*   n_nodes = (const int*)  d_in[1];
    /* d_in[2] = n_edges: unused by the reference */
    const float* gumbel  = (const float*)d_in[3];
    const float* emb     = (const float*)d_in[4];
    const float* pe      = (const float*)d_in[5];
    const float* W_pre   = (const float*)d_in[6];
    const float* b_pre   = (const float*)d_in[7];
    const float* Wih0    = (const float*)d_in[8];
    const float* Whh0    = (const float*)d_in[9];
    const float* bih0    = (const float*)d_in[10];
    const float* bhh0    = (const float*)d_in[11];
    const float* Wih1    = (const float*)d_in[12];
    const float* Whh1    = (const float*)d_in[13];
    const float* bih1    = (const float*)d_in[14];
    const float* bhh1    = (const float*)d_in[15];
    const float* node_W  = (const float*)d_in[16];
    const float* adj_W1  = (const float*)d_in[17];
    const float* adj_b1  = (const float*)d_in[18];
    const float* adj_W2  = (const float*)d_in[19];
    const float* adj_b2  = (const float*)d_in[20];

    char* base = (char*)d_ws;
    size_t off = 0;
    auto alloc = [&](size_t bytes) -> void* {
        void* p = base + off;
        off += (bytes + 255) & ~(size_t)255;
        return p;
    };

    _Float16* Wpre_h  = (_Float16*)alloc((size_t)256  * 512 * 2);
    _Float16* Wih0_h  = (_Float16*)alloc((size_t)1536 * 256 * 2);
    _Float16* Whh0_h  = (_Float16*)alloc((size_t)1536 * 512 * 2);
    _Float16* Wih1_h  = (_Float16*)alloc((size_t)1536 * 512 * 2);
    _Float16* Whh1_h  = (_Float16*)alloc((size_t)1536 * 512 * 2);
    _Float16* nodeW_h = (_Float16*)alloc((size_t)512  * 512 * 2);
    _Float16* W1a_h   = (_Float16*)alloc((size_t)512  * 512 * 2);
    _Float16* W1b_h   = (_Float16*)alloc((size_t)512  * 512 * 2);

    _Float16* A0      = (_Float16*)alloc((size_t)SEQ * 512 * 2);
    float*    seq_f32 = (float*)   alloc((size_t)SEQ * 256 * 4);
    _Float16* seq_h   = (_Float16*)alloc((size_t)SEQ * 256 * 2);
    float*    gi      = (float*)   alloc((size_t)SEQ * 1536 * 4);   // reused L0/L1
    _Float16* out1_h  = (_Float16*)alloc((size_t)SEQ * 512 * 2);
    _Float16* out2_h  = (_Float16*)alloc((size_t)SEQ * 512 * 2);
    _Float16* node_h  = (_Float16*)alloc((size_t)SEQ * 512 * 2);
    float*    Ubuf    = (float*)   alloc((size_t)SEQ * 512 * 4);
    float*    Vbuf    = (float*)   alloc((size_t)SEQ * 512 * 4);
    float*    gh      = (float*)   alloc((size_t)BATCH * 1536 * 4);
    float*    h_f32   = (float*)   alloc((size_t)BATCH * HIDDEN * 4);
    _Float16* h_h     = (_Float16*)alloc((size_t)BATCH * HIDDEN * 2);

    const int TB = 256;
    #define CONV(src, dst, n) k_convert<<<cdiv((n), TB), TB, 0, stream>>>((src), (dst), (n))
    CONV(W_pre,  Wpre_h,  256 * 512);
    CONV(Wih0,   Wih0_h,  1536 * 256);
    CONV(Whh0,   Whh0_h,  1536 * 512);
    CONV(Wih1,   Wih1_h,  1536 * 512);
    CONV(Whh1,   Whh1_h,  1536 * 512);
    CONV(node_W, nodeW_h, 512 * 512);
    #undef CONV
    k_split_w1<<<cdiv(512 * 512, TB), TB, 0, stream>>>(adj_W1, W1a_h, W1b_h);

    // big batched GEMMs: 32x64 strips per wave; per-step GRU GEMM: 16x16 tiles
    #define GEMM_BIG(A_, B_, bias_, C_, M_, N_, K_, half_)                          \
        do {                                                                        \
            int tiles_ = ((M_) / 32) * ((N_) / 64);                                 \
            k_wmma_gemm<2, 4><<<cdiv(tiles_, 4), 128, 0, stream>>>(                 \
                (A_), (B_), (bias_), (C_), (M_), (N_), (K_), (half_));              \
        } while (0)
    #define GEMM_STEP(A_, B_, bias_, C_, M_, N_, K_)                                \
        do {                                                                        \
            int tiles_ = ((M_) / 16) * ((N_) / 16);                                 \
            k_wmma_gemm<1, 1><<<cdiv(tiles_, 4), 128, 0, stream>>>(                 \
                (A_), (B_), (bias_), (C_), (M_), (N_), (K_), 0);                    \
        } while (0)

    // seq = [z+pe | emb] @ W_pre^T + b_pre, then mask -> f16
    k_build_a0<<<cdiv(SEQ * 512, TB), TB, 0, stream>>>(z, pe, emb, A0);
    GEMM_BIG(A0, Wpre_h, b_pre, seq_f32, SEQ, 256, 512, 0);
    k_mask_seq<<<cdiv(SEQ * 256, TB), TB, 0, stream>>>(seq_f32, seq_h, n_nodes);

    // -------------------- GRU layer 0 --------------------
    GEMM_BIG(seq_h, Wih0_h, bih0, gi, SEQ, 1536, 256, 0);
    k_zero_u32<<<cdiv(BATCH * HIDDEN, TB), TB, 0, stream>>>((unsigned int*)h_f32, BATCH * HIDDEN);
    k_zero_u32<<<cdiv(BATCH * HIDDEN / 2, TB), TB, 0, stream>>>((unsigned int*)h_h, BATCH * HIDDEN / 2);
    for (int t = 0; t < MAXN; ++t) {
        GEMM_STEP(h_h, Whh0_h, bhh0, gh, BATCH, 1536, 512);
        k_gru_gate<<<cdiv(BATCH * HIDDEN, TB), TB, 0, stream>>>(gi, gh, h_f32, h_h,
                                                                out1_h, n_nodes, t, 0);
    }

    // -------------------- GRU layer 1 --------------------
    GEMM_BIG(out1_h, Wih1_h, bih1, gi, SEQ, 1536, 512, 0);
    k_zero_u32<<<cdiv(BATCH * HIDDEN, TB), TB, 0, stream>>>((unsigned int*)h_f32, BATCH * HIDDEN);
    k_zero_u32<<<cdiv(BATCH * HIDDEN / 2, TB), TB, 0, stream>>>((unsigned int*)h_h, BATCH * HIDDEN / 2);
    for (int t = 0; t < MAXN; ++t) {
        GEMM_STEP(h_h, Whh1_h, bhh1, gh, BATCH, 1536, 512);
        k_gru_gate<<<cdiv(BATCH * HIDDEN, TB), TB, 0, stream>>>(gi, gh, h_f32, h_h,
                                                                out2_h, n_nodes, t, 1);
    }

    // node_emb = masked_rnn_out @ node_W^T  (direct f16 store)
    GEMM_BIG(out2_h, nodeW_h, nullptr, node_h, SEQ, 512, 512, 1);
    // U = node_emb @ W1a^T ; V = node_emb @ W1b^T + b1  (factored pair-MLP)
    GEMM_BIG(node_h, W1a_h, nullptr, Ubuf, SEQ, 512, 512, 0);
    GEMM_BIG(node_h, W1b_h, adj_b1, Vbuf, SEQ, 512, 512, 0);

    #undef GEMM_BIG
    #undef GEMM_STEP

    // adjacency output: zero (diagonal / untouched), then fused pair head
    k_zero_u32<<<cdiv(BATCH * MAXN * MAXN, TB), TB, 0, stream>>>((unsigned int*)d_out,
                                                                BATCH * MAXN * MAXN);
    k_pair_adj<<<cdiv(BATCH * NPAIR, 4), 128, 0, stream>>>(Ubuf, Vbuf, adj_W2, adj_b2,
                                                           gumbel, n_nodes, (float*)d_out);
}
